// DelayLineRNN_4286377361419
// MI455X (gfx1250) — compile-verified, matrix operand order checked by reading, not run on hardware
//
#include <hip/hip_runtime.h>
#include <hip/hip_bf16.h>
#include <math.h>

// ---------- types ----------
typedef __bf16 bf16_t;
typedef bf16_t v16bf __attribute__((ext_vector_type(16)));
typedef bf16_t v8bf  __attribute__((ext_vector_type(8)));
typedef bf16_t v4bf  __attribute__((ext_vector_type(4)));
typedef float  v8f   __attribute__((ext_vector_type(8)));

#define B_  64
#define T_  1024
#define I_  1024
#define H_  1024

__device__ __forceinline__ float fast_tanh(float x) {
#if __has_builtin(__builtin_amdgcn_tanhf)
    return __builtin_amdgcn_tanhf(x);     // native V_TANH_F32 (CDNA5 trans op)
#else
    return tanhf(x);
#endif
}

// ---------------------------------------------------------------------------
// Kernel 0a: convert weights to bf16, fuse biases (b_ih + b_hh)
// ---------------------------------------------------------------------------
__global__ void dlrnn_prep(const float* __restrict__ Wih, const float* __restrict__ Whh,
                           const float* __restrict__ bih, const float* __restrict__ bhh,
                           bf16_t* __restrict__ Wih_b, bf16_t* __restrict__ Whh_b,
                           float* __restrict__ bias) {
    const size_t NW = (size_t)H_ * I_;
    size_t stride = (size_t)gridDim.x * blockDim.x;
    for (size_t idx = (size_t)blockIdx.x * blockDim.x + threadIdx.x;
         idx < 2 * NW + H_; idx += stride) {
        if (idx < NW)            Wih_b[idx]        = (bf16_t)Wih[idx];
        else if (idx < 2 * NW)   Whh_b[idx - NW]   = (bf16_t)Whh[idx - NW];
        else                     bias[idx - 2*NW]  = bih[idx - 2*NW] + bhh[idx - 2*NW];
    }
}

// ---------------------------------------------------------------------------
// Kernel 0b: convert x (fp32, 256 MB) to bf16 once (pure HBM-stream pass),
// so the GEMM inner loop carries no v_cvt traffic and half the load bytes.
// ---------------------------------------------------------------------------
__global__ void dlrnn_cvt_x(const float* __restrict__ x, bf16_t* __restrict__ x_b) {
    const size_t N4 = (size_t)B_ * T_ * I_ / 4;
    size_t stride = (size_t)gridDim.x * blockDim.x;
    for (size_t i = (size_t)blockIdx.x * blockDim.x + threadIdx.x; i < N4; i += stride) {
        float4 v = ((const float4*)x)[i];
        v4bf o;
        o[0] = (bf16_t)v.x; o[1] = (bf16_t)v.y;
        o[2] = (bf16_t)v.z; o[3] = (bf16_t)v.w;
        ((v4bf*)x_b)[i] = o;
    }
}

// ---------------------------------------------------------------------------
// Fragment helper (CDNA5 16-bit A 16x32 / B 32x16 lane striping):
//   lane l (l<16):  row/col = l,     K runs {k0..k0+7, k0+16..k0+23}, k0 = kk
//   lane l (l>=16): row/col = l-16,  same runs with k0 = kk+8
// The two v8bf loads land in consecutive VGPRs -> folds into the v16bf operand.
// ---------------------------------------------------------------------------
__device__ __forceinline__ v16bf frag_from_bf16(const bf16_t* p, int k0) {
    v8bf lo = *(const v8bf*)(p + k0);
    v8bf hi = *(const v8bf*)(p + k0 + 16);
    return __builtin_shufflevector(lo, hi, 0,1,2,3,4,5,6,7,8,9,10,11,12,13,14,15);
}

// ---------------------------------------------------------------------------
// Kernel 1: xw[t][b][h] = x[b][t][:] @ W_ih[h][:] + bias[h]
// M = T*B = 65536 (row m -> t = m>>6, b = m&63), N = H, K = I.
// Block: 256 thr = 8 waves, tile 128(M) x 64(N); wave tile 64x16 (4 subtiles,
// i.e. 4 independent WMMA accumulator chains). Inner loop: b128 loads + WMMA.
// ---------------------------------------------------------------------------
__global__ __launch_bounds__(256) void dlrnn_xw_gemm(
        const bf16_t* __restrict__ x_b, const bf16_t* __restrict__ Wih_b,
        const float* __restrict__ bias, float* __restrict__ xw) {
    const int lane = threadIdx.x & 31;
    const int w    = threadIdx.x >> 5;     // 0..7
    const int half = lane >> 4;            // 0/1
    const int lm   = lane & 15;
    const int wm   = w & 1;                // M half of block tile
    const int wn   = w >> 1;               // 0..3 -> N slice
    const int m_wave = blockIdx.y * 128 + wm * 64;
    const int n      = blockIdx.x * 64 + wn * 16 + lm;   // this lane's N column

    const bf16_t* bp = Wih_b + (size_t)n * I_;
    const float   bv = bias[n];

    const bf16_t* ap[4];
#pragma unroll
    for (int i = 0; i < 4; ++i) {
        int m = m_wave + 16 * i + lm;      // A row for this lane / subtile
        int t = m >> 6, b = m & 63;
        ap[i] = x_b + ((size_t)b * T_ + t) * I_;
    }

    v8f acc[4] = {};
    for (int kk = 0; kk < I_; kk += 32) {
        const int k0 = kk + half * 8;
        v16bf bfrag = frag_from_bf16(bp, k0);
#pragma unroll
        for (int i = 0; i < 4; ++i) {
            v16bf afrag = frag_from_bf16(ap[i], k0);
            acc[i] = __builtin_amdgcn_wmma_f32_16x16x32_bf16(
                false, afrag, false, bfrag, (short)0, acc[i], false, false);
        }
    }

#pragma unroll
    for (int i = 0; i < 4; ++i) {
        const int mbase = m_wave + 16 * i + half * 8;
#pragma unroll
        for (int r = 0; r < 8; ++r) {
            const int m = mbase + r;
            xw[(size_t)m * H_ + n] = acc[i][r] + bv;
        }
    }
}

// ---------------------------------------------------------------------------
// Kernel 2: persistent recurrent scan.
// Grid: 128 blocks x 64 thr (2 waves). Block covers 32(M=batch) x 16(N=h);
// wave owns one 16x16 tile for all T steps. W_hh slice cached in LDS once.
// h history: 3-deep bf16 ring in ws -> one grid barrier per step.
// K reduction split over 4 independent WMMA accumulator chains to hide the
// matrix-pipe latency on the serial critical path; summed in f32 at the end.
// ---------------------------------------------------------------------------
__global__ __launch_bounds__(64) void dlrnn_scan(
        const float* __restrict__ xw, const bf16_t* __restrict__ Whh_b,
        float* __restrict__ states, float* __restrict__ hfinal,
        bf16_t* __restrict__ ring, unsigned* __restrict__ cnt) {
    __shared__ __align__(16) bf16_t Wsl[16 * H_];   // 32 KB: N-slice of W_hh

    const int lane = threadIdx.x & 31;
    const int w    = threadIdx.x >> 5;     // 0/1
    const int half = lane >> 4;
    const int lm   = lane & 15;
    const int n_base = (blockIdx.x & 63) * 16;
    const int m_wave = (blockIdx.x >> 6) * 32 + w * 16;

    {   // preload this block's 16 x 1024 W_hh slice (contiguous rows) into LDS
        const uint4* src = (const uint4*)(Whh_b + (size_t)n_base * H_);
        uint4* dst = (uint4*)Wsl;
        for (int i = threadIdx.x; i < (16 * H_ * 2) / 16; i += blockDim.x) dst[i] = src[i];
    }
    __syncthreads();

    const int n = n_base + lm;
    const bf16_t* wp  = Wsl + (size_t)lm * H_;      // B-matrix row (output col)
    const int arow    = m_wave + lm;                // A row = batch index
    const unsigned nblocks = gridDim.x;
    const bf16_t hhalf = (bf16_t)0.5f;

    for (int t = 0; t < T_; ++t) {
        const int s1 = (t + 2) % 3;                 // h_{t-1}
        const int s2 = (t + 1) % 3;                 // h_{t-2}
        const int sw = t % 3;                       // write slot
        const bf16_t* r1 = ring + (size_t)s1 * B_ * H_ + (size_t)arow * H_;
        const bf16_t* r2 = ring + (size_t)s2 * B_ * H_ + (size_t)arow * H_;
        const float* xwt = xw + (size_t)t * B_ * H_;
        const int mb = m_wave + half * 8;

        if (t + 1 < T_) {   // CDNA5 global_prefetch for next step's xw tile
            __builtin_prefetch(xw + (size_t)(t + 1) * B_ * H_ +
                               (size_t)mb * H_ + n, 0, 3);
        }

        // hoist the strided xw reads so they overlap the K reduction
        float xwv[8];
#pragma unroll
        for (int r = 0; r < 8; ++r) xwv[r] = xwt[(size_t)(mb + r) * H_ + n];

        v8f acc[4] = {};
        for (int kk = 0; kk < H_; kk += 128) {
#pragma unroll
            for (int c = 0; c < 4; ++c) {           // 4 independent chains
                const int k0 = kk + c * 32 + half * 8;
                v16bf bfrag = frag_from_bf16(wp, k0);   // ds_load_b128 x2
                // A = 0.5*(h_{t-1} + h_{t-2}) in packed bf16 (v_pk_add/mul)
                v8bf a1 = *(const v8bf*)(r1 + k0);
                v8bf a2 = *(const v8bf*)(r2 + k0);
                v8bf a3 = *(const v8bf*)(r1 + k0 + 16);
                v8bf a4 = *(const v8bf*)(r2 + k0 + 16);
                v8bf lo = (a1 + a2) * hhalf;
                v8bf hi = (a3 + a4) * hhalf;
                v16bf afrag = __builtin_shufflevector(lo, hi,
                    0,1,2,3,4,5,6,7,8,9,10,11,12,13,14,15);
                acc[c] = __builtin_amdgcn_wmma_f32_16x16x32_bf16(
                    false, afrag, false, bfrag, (short)0, acc[c], false, false);
            }
        }
        v8f accs = (acc[0] + acc[1]) + (acc[2] + acc[3]);

        // epilogue: add xw, tanh, write states + ring (+ h_final at t = T-1)
#pragma unroll
        for (int r = 0; r < 8; ++r) {
            const int b = mb + r;
            float v = fast_tanh(accs[r] + xwv[r]);
            states[((size_t)b * T_ + t) * H_ + n] = v;
            ring[(size_t)sw * B_ * H_ + (size_t)b * H_ + n] = (bf16_t)v;
            if (t == T_ - 1) hfinal[(size_t)b * H_ + n] = v;
        }

        // device-wide barrier (monotonic counter; slots never collide in-step)
        __threadfence();
        __syncthreads();
        if (threadIdx.x == 0) {
            __hip_atomic_fetch_add(cnt, 1u, __ATOMIC_RELEASE, __HIP_MEMORY_SCOPE_AGENT);
            const unsigned target = (unsigned)(t + 1) * nblocks;
            while (__hip_atomic_load(cnt, __ATOMIC_ACQUIRE, __HIP_MEMORY_SCOPE_AGENT) < target) {
                __builtin_amdgcn_s_sleep(2);
            }
        }
        __syncthreads();
    }
}

// ---------------------------------------------------------------------------
// Host launcher
// ---------------------------------------------------------------------------
extern "C" void kernel_launch(void* const* d_in, const int* in_sizes, int n_in,
                              void* d_out, int out_size, void* d_ws, size_t ws_size,
                              hipStream_t stream) {
    const float* x    = (const float*)d_in[0];
    const float* Wih  = (const float*)d_in[1];
    const float* Whh  = (const float*)d_in[2];
    const float* bih  = (const float*)d_in[3];
    const float* bhh  = (const float*)d_in[4];

    float* states = (float*)d_out;                                   // [B,T,H]
    float* hfinal = (float*)d_out + (size_t)B_ * T_ * H_;            // [1,B,H]

    // workspace carve-out
    char* ws = (char*)d_ws;
    const size_t XW_BYTES   = (size_t)T_ * B_ * H_ * sizeof(float);  // 256 MB
    const size_t XB_BYTES   = (size_t)B_ * T_ * I_ * sizeof(bf16_t); // 128 MB
    const size_t W_BYTES    = (size_t)H_ * I_ * sizeof(bf16_t);      // 2 MB each
    const size_t BIAS_BYTES = (size_t)H_ * sizeof(float);
    const size_t RING_BYTES = (size_t)3 * B_ * H_ * sizeof(bf16_t);  // 384 KB

    size_t off = 0;
    float*  xw    = (float*) (ws + off); off += XW_BYTES;
    bf16_t* x_b   = (bf16_t*)(ws + off); off += XB_BYTES;
    bf16_t* Wih_b = (bf16_t*)(ws + off); off += W_BYTES;
    bf16_t* Whh_b = (bf16_t*)(ws + off); off += W_BYTES;
    float*  bias  = (float*) (ws + off); off += BIAS_BYTES;
    bf16_t* ring  = (bf16_t*)(ws + off); off += RING_BYTES;
    unsigned* cnt = (unsigned*)(ws + off);

    // 0) weight/x conversion + bias fuse; deterministic ring/counter init
    hipLaunchKernelGGL(dlrnn_prep, dim3(2048), dim3(256), 0, stream,
                       Wih, Whh, bih, bhh, Wih_b, Whh_b, bias);
    hipLaunchKernelGGL(dlrnn_cvt_x, dim3(4096), dim3(256), 0, stream, x, x_b);
    hipMemsetAsync(ring, 0, RING_BYTES, stream);
    hipMemsetAsync(cnt, 0, sizeof(unsigned), stream);

    // 1) input-projection GEMM: grid = (N/64, M/128) = (16, 512)
    hipLaunchKernelGGL(dlrnn_xw_gemm, dim3(16, 512), dim3(256), 0, stream,
                       x_b, Wih_b, bias, xw);

    // 2) persistent recurrent scan: 128 co-resident blocks x 64 threads
    hipLaunchKernelGGL(dlrnn_scan, dim3(128), dim3(64), 0, stream,
                       xw, Whh_b, states, hfinal, ring, cnt);
}
